// ARGNNModel_41008347743020
// MI455X (gfx1250) — matrix operand order repaired and checked by reading, not compiled, not run on hardware
//
#include <hip/hip_runtime.h>
#include <hip/hip_bf16.h>
#include <cstdint>

// ---------- types ----------
typedef __bf16 bf16;
typedef __attribute__((ext_vector_type(16))) __bf16 v16bf;
typedef __attribute__((ext_vector_type(8)))  __bf16 v8bf;
typedef __attribute__((ext_vector_type(8)))  float  v8f;

static __device__ __forceinline__ bf16 f2bf(float f) {
  union { float f; unsigned u; } a; a.f = f;
  unsigned r = a.u + (0x7FFFu + ((a.u >> 16) & 1u));   // round-to-nearest-even
  union { unsigned short s; bf16 h; } o; o.s = (unsigned short)(r >> 16);
  return o.h;
}

static __device__ __forceinline__ float wred_sum(float v) {
#pragma unroll
  for (int m = 16; m; m >>= 1) v += __shfl_xor(v, m, 32);
  return v;
}
static __device__ __forceinline__ float wred_max(float v) {
#pragma unroll
  for (int m = 16; m; m >>= 1) v = fmaxf(v, __shfl_xor(v, m, 32));
  return v;
}

// ---------- elementwise utility kernels ----------
__global__ void k_fill(float* __restrict__ p, float v, long n) {
  long i = (long)blockIdx.x * blockDim.x + threadIdx.x;
  if (i < n) p[i] = v;
}
__global__ void k_copy(float* __restrict__ d, const float* __restrict__ s, long n) {
  long i = (long)blockIdx.x * blockDim.x + threadIdx.x;
  if (i < n) d[i] = s[i];
}
__global__ void k_f2b(bf16* __restrict__ d, const float* __restrict__ s, long n) {
  long i = (long)blockIdx.x * blockDim.x + threadIdx.x;
  if (i < n) d[i] = f2bf(s[i]);
}
// convert + transpose weight: W[K x Ncol] f32 -> Wt[Ncol x K] bf16
__global__ void k_wt(const float* __restrict__ W, bf16* __restrict__ Wt, int K, int Ncol) {
  int i = blockIdx.x * blockDim.x + threadIdx.x;
  if (i >= K * Ncol) return;
  int k = i / Ncol, n = i % Ncol;
  Wt[(long)n * K + k] = f2bf(W[i]);
}

// ---------- WMMA GEMM: C[M x Ncol] = A[M x K](bf16) * B[K x Ncol]; B given transposed Bt[Ncol x K] ----------
// One wave computes one 16x16 tile with v_wmma_f32_16x16x32_bf16.
// A fragment (16x32 bf16, ISA 7.12.2): lane<16 holds row M=lane, elems 0..7 = K0..7, 8..15 = K16..23;
// lane>=16 holds row M=lane-16, elems 0..7 = K8..15, 8..15 = K24..31.
// B fragment (32x16): lane&15 = column; same K order as A (lane<16: K0..7,K16..23; lane>=16: K8..15,K24..31).
__global__ __launch_bounds__(256)
void k_gemm(const bf16* __restrict__ A, const bf16* __restrict__ Bt,
            float* __restrict__ C, int K, int Ncol) {
  const int lane = threadIdx.x & 31;
  const int tm = blockIdx.x;
  const int tn = blockIdx.y * blockDim.y + threadIdx.y;
  if (tn * 16 >= Ncol) return;

  const int l15   = lane & 15;
  const int hi    = lane >> 4;        // 0 or 1
  const int khA   = hi << 3;          // +0 / +8
  const int kbB   = hi << 4;          // +0 / +16
  const bf16* arow = A  + (long)(tm * 16 + l15) * K;
  const bf16* brow = Bt + (long)(tn * 16 + l15) * K;

  v8f acc = {};
  for (int k0 = 0; k0 < K; k0 += 32) {
    union { v16bf v; v8bf h[2]; } a, b;
    a.h[0] = *(const v8bf*)(arow + k0 + khA);
    a.h[1] = *(const v8bf*)(arow + k0 + 16 + khA);
    b.h[0] = *(const v8bf*)(brow + k0 + khA);
    b.h[1] = *(const v8bf*)(brow + k0 + 16 + khA);
    if (k0 + 32 < K) __builtin_prefetch(arow + k0 + 32, 0, 1);  // global_prefetch_b8
    acc = __builtin_amdgcn_wmma_f32_16x16x32_bf16(
        /*neg_a=*/false, a.v, /*neg_b=*/false, b.v,
        /*c_mod=*/(short)0, acc, /*reuse_a=*/false, /*reuse_b=*/false);
  }
  // C/D layout: VGPR v -> row tm*16 + v + 8*hi, column tn*16 + (lane&15)
  float* cp = C + (long)(tm * 16 + (hi << 3)) * Ncol + tn * 16 + l15;
#pragma unroll
  for (int v = 0; v < 8; ++v) cp[(long)v * Ncol] = acc[v];
}

// ---------- fused bias (+add) + LayerNorm (+ReLU), wave per row; writes f32 and optional bf16 ----------
__global__ __launch_bounds__(256)
void k_ln(const float* __restrict__ A, const float* __restrict__ Badd,
          const float* __restrict__ bias, const float* __restrict__ lg,
          const float* __restrict__ lb, float* __restrict__ o32,
          bf16* __restrict__ o16, int N, int D, int relu) {
  int r = blockIdx.x * (blockDim.x >> 5) + (threadIdx.x >> 5);
  if (r >= N) return;
  const int lane = threadIdx.x & 31;
  const int per = D >> 5;            // D in {64,128} -> 2 or 4
  const float* a = A + (long)r * D;
  const float* b2 = Badd ? Badd + (long)r * D : nullptr;
  float v[4]; float s = 0.f;
#pragma unroll 4
  for (int k = 0; k < per; ++k) {
    int c = lane * per + k;
    float t = a[c];
    if (b2)  t += b2[c];
    if (bias) t += bias[c];
    v[k] = t; s += t;
  }
  float mean = wred_sum(s) * (1.0f / D);
  float q = 0.f;
#pragma unroll 4
  for (int k = 0; k < per; ++k) { v[k] -= mean; q += v[k] * v[k]; }
  float inv = rsqrtf(wred_sum(q) * (1.0f / D) + 1e-5f);
#pragma unroll 4
  for (int k = 0; k < per; ++k) {
    int c = lane * per + k;
    float o = v[k] * inv * lg[c] + lb[c];
    if (relu) o = fmaxf(o, 0.f);
    o32[(long)r * D + c] = o;
    if (o16) o16[(long)r * D + c] = f2bf(o);
  }
}

// ---------- metric head: g = clip(softplus(2*(raw+bias))/2, MIN_G, MAX_G), elementwise over N x 128 ----------
__global__ void k_softplus(float* __restrict__ gio, const float* __restrict__ bias, long n) {
  long i = (long)blockIdx.x * blockDim.x + threadIdx.x;
  if (i >= n) return;
  int c = (int)(i & 127);
  float t = 2.f * (gio[i] + bias[c]);
  float sp = 0.5f * (fmaxf(t, 0.f) + log1pf(expf(-fabsf(t))));
  gio[i] = fminf(fmaxf(sp, 0.001f), 10.0f);
}

// ---------- degree: deg prefilled with 1.0 (self loop), +1 per incoming edge ----------
__global__ void k_deg(const int* __restrict__ col, float* __restrict__ deg, int E) {
  int e = blockIdx.x * blockDim.x + threadIdx.x;
  if (e < E) atomicAdd(&deg[col[e]], 1.0f);
}

// ---------- neighbor scatter-sum: sum[col] += h[row]; wave per edge, float4 per lane (D=128) ----------
__global__ __launch_bounds__(256)
void k_scatter(const int* __restrict__ row, const int* __restrict__ col,
               const float* __restrict__ h, float* __restrict__ sum, int E) {
  int e = blockIdx.x * (blockDim.x >> 5) + (threadIdx.x >> 5);
  if (e >= E) return;
  int lane = threadIdx.x & 31;
  int j = row[e], i = col[e];
  float4 v = ((const float4*)(h + (long)j * 128))[lane];
  float* d = sum + (long)i * 128 + lane * 4;
  atomicAdd(d + 0, v.x); atomicAdd(d + 1, v.y);
  atomicAdd(d + 2, v.z); atomicAdd(d + 3, v.w);
}

// ---------- build concat(h, sum/deg) in bf16: cat[N x 256] ----------
__global__ void k_cat(const float* __restrict__ h, const float* __restrict__ sum,
                      const float* __restrict__ deg, bf16* __restrict__ cat, long n) {
  long i = (long)blockIdx.x * blockDim.x + threadIdx.x;   // over N*128
  if (i >= n) return;
  long node = i >> 7; int c = (int)(i & 127);
  cat[node * 256 + c]       = f2bf(h[i]);
  cat[node * 256 + 128 + c] = f2bf(sum[i] / deg[node]);
}

// ---------- edge messages (din=128): wave per edge; tau/alpha via shfl reductions; atomic scatter ----------
__global__ __launch_bounds__(256)
void k_edge(const int* __restrict__ row, const int* __restrict__ col,
            const float* __restrict__ x, const float* __restrict__ g,
            const float* __restrict__ xw, float* __restrict__ agg, int E, int dout) {
  int e = blockIdx.x * (blockDim.x >> 5) + (threadIdx.x >> 5);
  if (e >= E) return;
  const int lane = threadIdx.x & 31;
  const int j = row[e], i = col[e];          // j = source, i = target
  float4 xi = ((const float4*)(x + (long)i * 128))[lane];
  float4 xj = ((const float4*)(x + (long)j * 128))[lane];
  float4 gi = ((const float4*)(g + (long)i * 128))[lane];

  float dx[4] = { xj.x - xi.x, xj.y - xi.y, xj.z - xi.z, xj.w - xi.w };
  float gg[4] = { gi.x, gi.y, gi.z, gi.w };
  float xa[4] = { xi.x, xi.y, xi.z, xi.w };
  float xb[4] = { xj.x, xj.y, xj.z, xj.w };

  float sd = 0.f, wi = 0.f, ni = 0.f, nj = 0.f;
#pragma unroll
  for (int c = 0; c < 4; ++c) {
    sd += dx[c] * dx[c];
    wi += gg[c] * xa[c] * xb[c];
    ni += gg[c] * xa[c] * xa[c];
    nj += gg[c] * xb[c] * xb[c];
  }
  sd = wred_sum(sd); wi = wred_sum(wi); ni = wred_sum(ni); nj = wred_sum(nj);

  float dn = fmaxf(sqrtf(sd), 1e-8f);
  float inv2 = 1.f / (dn * dn);
  // tanh(-log g) == (1-g^2)/(1+g^2) for g>0 (g clipped to [1e-3, 10])
  float tt = 0.f;
#pragma unroll
  for (int c = 0; c < 4; ++c) {
    float g2 = gg[c] * gg[c];
    tt += dx[c] * dx[c] * ((1.f - g2) / (1.f + g2));
  }
  float tau = wred_sum(tt) * inv2;
  float alpha = 1.f / (1.f + expf(-(wi / (sqrtf(ni * nj) + 1e-8f))));
  float s = tau * alpha;

  const float* xwj = xw + (long)j * dout;
  float* ai = agg + (long)i * dout;
  for (int c = lane; c < dout; c += 32) atomicAdd(&ai[c], s * xwj[c]);
}

// ---------- final log_softmax over D=64, wave per row ----------
__global__ __launch_bounds__(256)
void k_lsm(const float* __restrict__ in, float* __restrict__ out, int N) {
  int r = blockIdx.x * (blockDim.x >> 5) + (threadIdx.x >> 5);
  if (r >= N) return;
  int lane = threadIdx.x & 31;
  float v0 = in[(long)r * 64 + lane * 2];
  float v1 = in[(long)r * 64 + lane * 2 + 1];
  float m = wred_max(fmaxf(v0, v1));
  float se = wred_sum(expf(v0 - m) + expf(v1 - m));
  float lse = m + logf(se);
  out[(long)r * 64 + lane * 2]     = v0 - lse;
  out[(long)r * 64 + lane * 2 + 1] = v1 - lse;
}

// ---------- host orchestration ----------
extern "C" void kernel_launch(void* const* d_in, const int* in_sizes, int n_in,
                              void* d_out, int out_size, void* d_ws, size_t ws_size,
                              hipStream_t stream) {
  const int N = in_sizes[0] / 256;       // 50000
  const int E = in_sizes[1] / 2;         // 800000

  const float* x   = (const float*)d_in[0];
  const int*   row = (const int*)d_in[1];        // edge_index[0] = sources
  const int*   colv = row + E;                   // edge_index[1] = targets
  const float* encW = (const float*)d_in[2];
  const float* encb = (const float*)d_in[3];
  const float* encg = (const float*)d_in[4];
  const float* encbb= (const float*)d_in[5];

  struct Layer {
    const float *m1W, *m1b, *m1g, *m1bb, *m2W, *m2b, *msgW, *selfW, *selfb, *lng, *lnb;
    int dout;
  } L[3];
  const int douts[3] = {128, 128, 64};
  for (int l = 0; l < 3; ++l) {
    int b = 6 + l * 11;
    L[l] = { (const float*)d_in[b+0], (const float*)d_in[b+1], (const float*)d_in[b+2],
             (const float*)d_in[b+3], (const float*)d_in[b+4], (const float*)d_in[b+5],
             (const float*)d_in[b+6], (const float*)d_in[b+7], (const float*)d_in[b+8],
             (const float*)d_in[b+9], (const float*)d_in[b+10], douts[l] };
  }

  // ---- workspace bump allocator ----
  char* ws = (char*)d_ws; size_t off = 0;
  auto alloc = [&](size_t bytes) -> void* {
    off = (off + 255) & ~(size_t)255;
    void* p = ws + off; off += bytes; return p;
  };
  float* h32   = (float*)alloc((size_t)N * 128 * 4);
  bf16*  h16   = (bf16*) alloc((size_t)N * 128 * 2);
  bf16*  cat16 = (bf16*) alloc((size_t)N * 256 * 2);   // also reused as x16 for encoder
  float* mid32 = (float*)alloc((size_t)N * 64 * 4);
  bf16*  mid16 = (bf16*) alloc((size_t)N * 64 * 2);
  float* g32   = (float*)alloc((size_t)N * 128 * 4);
  float* xw32  = (float*)alloc((size_t)N * 128 * 4);
  float* acc32 = (float*)alloc((size_t)N * 128 * 4);   // neighbor-sum, then msg agg
  float* sf32  = (float*)alloc((size_t)N * 128 * 4);   // self GEMM out / encoder tmp
  float* deg   = (float*)alloc((size_t)N * 4);
  bf16*  encWt = (bf16*) alloc(256 * 128 * 2);
  bf16*  m1Wt[3], *m2Wt[3], *msgWt[3], *selfWt[3];
  for (int l = 0; l < 3; ++l) {
    m1Wt[l]   = (bf16*)alloc(256 * 64 * 2);
    m2Wt[l]   = (bf16*)alloc(64 * 128 * 2);
    msgWt[l]  = (bf16*)alloc(128 * (size_t)douts[l] * 2);
    selfWt[l] = (bf16*)alloc(128 * (size_t)douts[l] * 2);
  }
  (void)ws_size; (void)n_in;

  auto grid1 = [](long n, int bs) { return dim3((unsigned)((n + bs - 1) / bs)); };
  auto wconv = [&](const float* W, bf16* Wt, int K, int Ncol) {
    k_wt<<<grid1((long)K * Ncol, 256), 256, 0, stream>>>(W, Wt, K, Ncol);
  };
  auto gemm = [&](const bf16* A, const bf16* Bt, float* C, int K, int Ncol) {
    int nt = Ncol / 16;
    int wy = nt < 8 ? nt : 8;
    dim3 blk(32, wy), grd(N / 16, nt / wy);
    k_gemm<<<grd, blk, 0, stream>>>(A, Bt, C, K, Ncol);
  };
  const dim3 eblk(256), egrd((E + 7) / 8);   // wave per edge, 8 waves/block
  const dim3 rgrd((N + 7) / 8);              // wave per row

  // ---- one-time weight convert/transpose ----
  wconv(encW, encWt, 256, 128);
  for (int l = 0; l < 3; ++l) {
    wconv(L[l].m1W,   m1Wt[l],   256, 64);
    wconv(L[l].m2W,   m2Wt[l],   64, 128);
    wconv(L[l].msgW,  msgWt[l],  128, L[l].dout);
    wconv(L[l].selfW, selfWt[l], 128, L[l].dout);
  }

  // ---- degree (self-loop contributes 1) ----
  k_fill<<<grid1(N, 256), 256, 0, stream>>>(deg, 1.0f, N);
  k_deg<<<grid1(E, 256), 256, 0, stream>>>(colv, deg, E);

  // ---- encoder: h = relu(LN(x @ W + b)) ----
  k_f2b<<<grid1((long)N * 256, 256), 256, 0, stream>>>(cat16, x, (long)N * 256);
  gemm(cat16, encWt, sf32, 256, 128);
  k_ln<<<rgrd, 256, 0, stream>>>(sf32, nullptr, encb, encg, encbb, h32, h16, N, 128, 1);

  // ---- layers ----
  for (int l = 0; l < 3; ++l) {
    const int dout = L[l].dout;
    // scatter-mean (self-loop: init sum = h)
    k_copy<<<grid1((long)N * 128, 256), 256, 0, stream>>>(acc32, h32, (long)N * 128);
    k_scatter<<<egrd, eblk, 0, stream>>>(row, colv, h32, acc32, E);
    k_cat<<<grid1((long)N * 128, 256), 256, 0, stream>>>(h32, acc32, deg, cat16, (long)N * 128);
    // metric MLP -> g
    gemm(cat16, m1Wt[l], mid32, 256, 64);
    k_ln<<<rgrd, 256, 0, stream>>>(mid32, nullptr, L[l].m1b, L[l].m1g, L[l].m1bb,
                                   mid32, mid16, N, 64, 1);
    gemm(mid16, m2Wt[l], g32, 64, 128);
    k_softplus<<<grid1((long)N * 128, 256), 256, 0, stream>>>(g32, L[l].m2b, (long)N * 128);
    // per-node projections
    gemm(h16, msgWt[l], xw32, 128, dout);
    gemm(h16, selfWt[l], sf32, 128, dout);
    // edge messages -> agg
    k_fill<<<grid1((long)N * dout, 256), 256, 0, stream>>>(acc32, 0.0f, (long)N * dout);
    k_edge<<<egrd, eblk, 0, stream>>>(row, colv, h32, g32, xw32, acc32, E, dout);
    // out = LN(agg + self + b); relu except last layer
    k_ln<<<rgrd, 256, 0, stream>>>(acc32, sf32, L[l].selfb, L[l].lng, L[l].lnb,
                                   h32, h16, N, dout, l < 2 ? 1 : 0);
  }

  // ---- log_softmax ----
  k_lsm<<<rgrd, 256, 0, stream>>>(h32, (float*)d_out, N);
  (void)out_size;
}
